// MultiHeadAttention_54554674593946
// MI455X (gfx1250) — matrix operand order repaired and checked by reading, not compiled, not run on hardware
//
#include <hip/hip_runtime.h>

typedef __attribute__((ext_vector_type(16))) _Float16 v16h;
typedef __attribute__((ext_vector_type(8)))  _Float16 v8h;
typedef __attribute__((ext_vector_type(8)))  float    v8f;
typedef __attribute__((ext_vector_type(4)))  int      i32x4;

#define WMMA_F16(a, b, c) \
  __builtin_amdgcn_wmma_f32_16x16x32_f16(false, (a), false, (b), (short)0, (c), false, false)

// ---- optional gfx1250 async global->LDS path (guarded; sync fallback) -----
#if defined(__has_builtin)
#  if __has_builtin(__builtin_amdgcn_global_load_async_to_lds_b128) && \
      __has_builtin(__builtin_amdgcn_s_wait_asynccnt)
#    define HAVE_ASYNC_LDS 1
#  endif
#endif
#ifndef HAVE_ASYNC_LDS
#  define HAVE_ASYNC_LDS 0
#endif

#if HAVE_ASYNC_LDS
static __device__ inline void async_copy16(_Float16* lds, const _Float16* g) {
  // src: AS1 (global) int32x4*, dst: AS3 (LDS) int32x4*, imm offset, imm cpol
  __attribute__((address_space(3))) _Float16* l3 =
      (__attribute__((address_space(3))) _Float16*)lds;          // addrspacecast
  __builtin_amdgcn_global_load_async_to_lds_b128(
      (__attribute__((address_space(1))) i32x4*)(unsigned long long)g,
      (__attribute__((address_space(3))) i32x4*)l3,
      0, 0);
}
#endif

// ---------------------------------------------------------------------------
// Fragment loaders for V_WMMA_F32_16X16X32_F16 (wave32).
// A (16x32, MxK) row-major, ld: lane L: row=L&15; halves0-7 = K(L<16?0..7:8..15),
//   halves8-15 = +16.  Two 16B loads.
// B (32x16, KxN) from an [N,K] row-major source: lane L: n=L&15;
//   halves0-7 = K(L<16?0..7:8..15), halves8-15 = +16. Two 16B loads.
// ---------------------------------------------------------------------------
static __device__ inline v16h load_frag_A(const _Float16* __restrict__ base, int ld) {
  const int lane = threadIdx.x & 31;
  const int row  = lane & 15;
  const int koff = (lane >> 4) << 3;  // 0 or 8
  const _Float16* p = base + (size_t)row * ld + koff;
  v8h lo = *(const v8h*)(p);
  v8h hi = *(const v8h*)(p + 16);
  v16h r;
#pragma unroll
  for (int i = 0; i < 8; ++i) { r[i] = lo[i]; r[i + 8] = hi[i]; }
  return r;
}

static __device__ inline v16h load_frag_B(const _Float16* __restrict__ base, int ld) {
  const int lane = threadIdx.x & 31;
  const int n    = lane & 15;
  const int koff = (lane >> 4) << 3;  // 0 or 8
  const _Float16* p = base + (size_t)n * ld + koff;
  v8h lo = *(const v8h*)(p);
  v8h hi = *(const v8h*)(p + 16);
  v16h r;
#pragma unroll
  for (int i = 0; i < 8; ++i) { r[i] = lo[i]; r[i + 8] = hi[i]; }
  return r;
}

static __device__ inline v16h frag_from_lds(const _Float16* row0, int pitch) {
  const int lane = threadIdx.x & 31;
  const int n    = lane & 15;
  const int koff = (lane >> 4) << 3;  // 0 or 8
  const _Float16* p = row0 + (size_t)n * pitch + koff;
  v8h lo = *(const v8h*)(p);
  v8h hi = *(const v8h*)(p + 16);
  v16h r;
#pragma unroll
  for (int i = 0; i < 8; ++i) { r[i] = lo[i]; r[i + 8] = hi[i]; }
  return r;
}

// ---------------------------------------------------------------------------
// Precision staging kernels.
// ---------------------------------------------------------------------------
__global__ __launch_bounds__(256) void k_f32_to_f16(const float* __restrict__ src,
                                                    _Float16* __restrict__ dst, int n) {
  int i = blockIdx.x * 256 + threadIdx.x;
  if (i < n) dst[i] = (_Float16)src[i];
}

__global__ __launch_bounds__(256) void k_transpose_f16(const float* __restrict__ W,
                                                       _Float16* __restrict__ WT, int D) {
  int i = blockIdx.x * 256 + threadIdx.x;  // i = k*D + n
  if (i < D * D) {
    int k = i / D, n = i % D;
    WT[(size_t)n * D + k] = (_Float16)W[i];
  }
}

// ---------------------------------------------------------------------------
// Projection GEMM: Y = A[M,K] @ Bt[N,K]^T, f16 in, f32 acc, f16 out.
// Block: 256 thr (8 waves) = 128 rows x 64 cols; wave tile 16x64.
// 64x32 B slab per K-step shared by all 8 waves -> staged in LDS, double-
// buffered, async global->LDS when available.
// mode 0: store [b,h,s,hd] (Q,K);  mode 1: store [b,h,hd,s] (V^T).
// ---------------------------------------------------------------------------
__global__ __launch_bounds__(256) void k_gemm_proj(
    const _Float16* __restrict__ A, const _Float16* __restrict__ Bt,
    _Float16* __restrict__ out, int M, int N, int K, int S, int mode) {
  const int tid  = threadIdx.x;
  const int lane = tid & 31;
  const int wave = tid >> 5;
  const int m0 = blockIdx.x * 128 + wave * 16;
  const int n0 = blockIdx.y * 64;

  __shared__ _Float16 Blds[2][64][40];  // 80B pitch -> every 16B chunk aligned

  const int srow = tid >> 2;
  const int scol = (tid & 3) << 3;
  const _Float16* gB = Bt + (size_t)(n0 + srow) * K + scol;

  auto stage = [&](int buf, int kk) {
#if HAVE_ASYNC_LDS
    async_copy16(&Blds[buf][srow][scol], gB + kk);
#else
    v8h t = *(const v8h*)(gB + kk);
    *(v8h*)&Blds[buf][srow][scol] = t;
#endif
  };

  const v8f z = {0.f, 0.f, 0.f, 0.f, 0.f, 0.f, 0.f, 0.f};
  v8f acc[4] = {z, z, z, z};

  stage(0, 0);
  v16h a_cur = load_frag_A(A + (size_t)m0 * K, K);

  int ib = 0;
  for (int kk = 0; kk < K; kk += 32, ib ^= 1) {
    const bool more = (kk + 32) < K;
    if (more) stage(ib ^ 1, kk + 32);                        // next B slab in flight
    v16h a_nxt = more ? load_frag_A(A + (size_t)m0 * K + kk + 32, K) : a_cur;
    if (kk + 256 < K)                                        // stream-ahead hint
      __builtin_prefetch(A + (size_t)(m0 + (lane & 15)) * K + kk + 256, 0, 0);
#if HAVE_ASYNC_LDS
    if (more) __builtin_amdgcn_s_wait_asynccnt(1);           // buf[ib] landed
    else      __builtin_amdgcn_s_wait_asynccnt(0);
#endif
    __syncthreads();
#pragma unroll
    for (int f = 0; f < 4; ++f) {
      v16h b = frag_from_lds(&Blds[ib][f * 16][0], 40);
      acc[f] = WMMA_F16(a_cur, b, acc[f]);
    }
    a_cur = a_nxt;
    __syncthreads();                                         // reads done before restage
  }

  const int rbase = m0 + ((lane >> 4) << 3);
  const int cbase = n0 + (lane & 15);
  (void)N;
#pragma unroll
  for (int f = 0; f < 4; ++f) {
#pragma unroll
    for (int j = 0; j < 8; ++j) {
      const int r = rbase + j;
      const int c = cbase + f * 16;
      const int bb = r / S, s = r % S;
      const int h = c >> 7, dd = c & 127;
      const size_t idx = (mode == 0)
          ? ((((size_t)(bb * 16 + h)) * S + s) * 128 + dd)
          : ((((size_t)(bb * 16 + h)) * 128 + dd) * S + s);
      out[idx] = (_Float16)acc[f][j];
    }
  }
}

// ---------------------------------------------------------------------------
// Output projection: out = ctx[M,K] @ Wo^T[N,K]^T + b_o, f32 out.
// ---------------------------------------------------------------------------
__global__ __launch_bounds__(256) void k_gemm_out(
    const _Float16* __restrict__ A, const _Float16* __restrict__ Bt,
    const float* __restrict__ bias, float* __restrict__ out,
    int M, int N, int K) {
  const int tid  = threadIdx.x;
  const int lane = tid & 31;
  const int wave = tid >> 5;
  const int m0 = blockIdx.x * 128 + wave * 16;
  const int n0 = blockIdx.y * 64;

  __shared__ _Float16 Blds[2][64][40];

  const int srow = tid >> 2;
  const int scol = (tid & 3) << 3;
  const _Float16* gB = Bt + (size_t)(n0 + srow) * K + scol;

  auto stage = [&](int buf, int kk) {
#if HAVE_ASYNC_LDS
    async_copy16(&Blds[buf][srow][scol], gB + kk);
#else
    v8h t = *(const v8h*)(gB + kk);
    *(v8h*)&Blds[buf][srow][scol] = t;
#endif
  };

  const v8f z = {0.f, 0.f, 0.f, 0.f, 0.f, 0.f, 0.f, 0.f};
  v8f acc[4] = {z, z, z, z};

  stage(0, 0);
  v16h a_cur = load_frag_A(A + (size_t)m0 * K, K);

  int ib = 0;
  for (int kk = 0; kk < K; kk += 32, ib ^= 1) {
    const bool more = (kk + 32) < K;
    if (more) stage(ib ^ 1, kk + 32);
    v16h a_nxt = more ? load_frag_A(A + (size_t)m0 * K + kk + 32, K) : a_cur;
    if (kk + 256 < K)
      __builtin_prefetch(A + (size_t)(m0 + (lane & 15)) * K + kk + 256, 0, 0);
#if HAVE_ASYNC_LDS
    if (more) __builtin_amdgcn_s_wait_asynccnt(1);
    else      __builtin_amdgcn_s_wait_asynccnt(0);
#endif
    __syncthreads();
#pragma unroll
    for (int f = 0; f < 4; ++f) {
      v16h b = frag_from_lds(&Blds[ib][f * 16][0], 40);
      acc[f] = WMMA_F16(a_cur, b, acc[f]);
    }
    a_cur = a_nxt;
    __syncthreads();
  }

  const int rbase = m0 + ((lane >> 4) << 3);
  const int cbase = n0 + (lane & 15);
#pragma unroll
  for (int f = 0; f < 4; ++f) {
    const int c = cbase + f * 16;
    const float bv = bias[c];
#pragma unroll
    for (int j = 0; j < 8; ++j) {
      out[(size_t)(rbase + j) * N + c] = acc[f][j] + bv;
    }
  }
}

// ---------------------------------------------------------------------------
// Flash attention: one wave (32-thread block) per (b, h, 16-query tile).
// 32 keys/iter, online softmax; V fragments loaded before the softmax VALU
// block so exp/shuffle work hides their latency.
// Q:[B,H,S,128] Km:[B,H,S,128] Vt:[B,H,128,S] ctx:[B,S,2048] (all f16)
// ---------------------------------------------------------------------------
__global__ __launch_bounds__(32) void k_attn(
    const _Float16* __restrict__ Q, const _Float16* __restrict__ Km,
    const _Float16* __restrict__ Vt, _Float16* __restrict__ ctx, int S) {
  const int lane   = threadIdx.x & 31;
  const int qtiles = S >> 4;
  const int bh = blockIdx.x / qtiles;
  const int qt = blockIdx.x % qtiles;
  const int q0 = qt << 4;
  const int b  = bh >> 4;  // H = 16
  const int h  = bh & 15;

  const _Float16* Qp = Q  + (size_t)bh * S * 128;
  const _Float16* Kp = Km + (size_t)bh * S * 128;
  const _Float16* Vp = Vt + (size_t)bh * 128 * S;

  __shared__ _Float16 Plds[16][40];

  v16h qa[4];
#pragma unroll
  for (int d4 = 0; d4 < 4; ++d4)
    qa[d4] = load_frag_A(Qp + (size_t)q0 * 128 + d4 * 32, 128);

  const v8f z = {0.f, 0.f, 0.f, 0.f, 0.f, 0.f, 0.f, 0.f};
  v8f o[8] = {z, z, z, z, z, z, z, z};
  float mrow[8], lrow[8];
#pragma unroll
  for (int j = 0; j < 8; ++j) { mrow[j] = -1e30f; lrow[j] = 0.f; }

  const float scale = 0.08838834764831845f;  // 1/sqrt(128)
  const int rsub = (lane >> 4) << 3;
  const int ncol = lane & 15;

  for (int k0 = 0; k0 <= q0 + 15; k0 += 32) {
    v16h kb[8];
#pragma unroll
    for (int d4 = 0; d4 < 4; ++d4) {
      kb[d4 * 2]     = load_frag_B(Kp + (size_t)k0 * 128 + d4 * 32, 128);
      kb[d4 * 2 + 1] = load_frag_B(Kp + (size_t)(k0 + 16) * 128 + d4 * 32, 128);
    }
    v8f s0 = z, s1 = z;
#pragma unroll
    for (int d4 = 0; d4 < 4; ++d4) {
      s0 = WMMA_F16(qa[d4], kb[d4 * 2], s0);
      s1 = WMMA_F16(qa[d4], kb[d4 * 2 + 1], s1);
    }

    // issue V fragment loads now; softmax below hides their latency
    v16h vb[8];
#pragma unroll
    for (int g = 0; g < 8; ++g)
      vb[g] = load_frag_B(Vp + (size_t)(g * 16) * S + k0, S);

    float p0[8], p1[8];
#pragma unroll
    for (int j = 0; j < 8; ++j) {
      const int q = q0 + rsub + j;
      float x0 = s0[j] * scale;
      float x1 = s1[j] * scale;
      if (k0 + ncol      > q) x0 = -1e30f;
      if (k0 + 16 + ncol > q) x1 = -1e30f;
      float t = fmaxf(x0, x1);
      t = fmaxf(t, __shfl_xor(t, 1, 32));
      t = fmaxf(t, __shfl_xor(t, 2, 32));
      t = fmaxf(t, __shfl_xor(t, 4, 32));
      t = fmaxf(t, __shfl_xor(t, 8, 32));
      const float mnew = fmaxf(mrow[j], t);
      const float fac  = __expf(mrow[j] - mnew);
      p0[j] = __expf(x0 - mnew);
      p1[j] = __expf(x1 - mnew);
      float rs = p0[j] + p1[j];
      rs += __shfl_xor(rs, 1, 32);
      rs += __shfl_xor(rs, 2, 32);
      rs += __shfl_xor(rs, 4, 32);
      rs += __shfl_xor(rs, 8, 32);
      lrow[j] = lrow[j] * fac + rs;
      mrow[j] = mnew;
#pragma unroll
      for (int g = 0; g < 8; ++g) o[g][j] *= fac;
    }

    // C-layout -> A-fragment layout via LDS bounce
#pragma unroll
    for (int j = 0; j < 8; ++j) {
      Plds[rsub + j][ncol]      = (_Float16)p0[j];
      Plds[rsub + j][16 + ncol] = (_Float16)p1[j];
    }
    __syncthreads();
    v16h pa;
    {
      const int r = lane & 15;
      v8h lo = *(const v8h*)&Plds[r][rsub];
      v8h hi = *(const v8h*)&Plds[r][16 + rsub];
#pragma unroll
      for (int i = 0; i < 8; ++i) { pa[i] = lo[i]; pa[i + 8] = hi[i]; }
    }
    __syncthreads();

#pragma unroll
    for (int g = 0; g < 8; ++g)
      o[g] = WMMA_F16(pa, vb[g], o[g]);
  }

#pragma unroll
  for (int j = 0; j < 8; ++j) {
    const float inv = 1.0f / lrow[j];
    const size_t row = (size_t)b * S + (size_t)(q0 + rsub + j);
#pragma unroll
    for (int g = 0; g < 8; ++g) {
      ctx[row * 2048 + h * 128 + g * 16 + ncol] = (_Float16)(o[g][j] * inv);
    }
  }
}

// ---------------------------------------------------------------------------
extern "C" void kernel_launch(void* const* d_in, const int* in_sizes, int n_in,
                              void* d_out, int out_size, void* d_ws, size_t ws_size,
                              hipStream_t stream) {
  (void)in_sizes; (void)n_in; (void)out_size; (void)ws_size;
  const float* x  = (const float*)d_in[0];
  const float* Wq = (const float*)d_in[1];
  const float* Wk = (const float*)d_in[2];
  const float* Wv = (const float*)d_in[3];
  const float* Wo = (const float*)d_in[4];
  const float* bo = (const float*)d_in[5];
  float* out = (float*)d_out;

  const int B = 2, S = 2048, D = 2048, H = 16;
  const int M = B * S;  // 4096

  // workspace (f16): xh | WqT | WkT | WvT | WoT | Q | K | Vt | ctx
  char* ws = (char*)d_ws;
  size_t off = 0;
  _Float16* xh   = (_Float16*)(ws + off); off += (size_t)M * D * 2;
  _Float16* WqT  = (_Float16*)(ws + off); off += (size_t)D * D * 2;
  _Float16* WkT  = (_Float16*)(ws + off); off += (size_t)D * D * 2;
  _Float16* WvT  = (_Float16*)(ws + off); off += (size_t)D * D * 2;
  _Float16* WoT  = (_Float16*)(ws + off); off += (size_t)D * D * 2;
  _Float16* Qh   = (_Float16*)(ws + off); off += (size_t)B * H * S * 128 * 2;
  _Float16* Kh   = (_Float16*)(ws + off); off += (size_t)B * H * S * 128 * 2;
  _Float16* Vt   = (_Float16*)(ws + off); off += (size_t)B * H * 128 * S * 2;
  _Float16* ctxh = (_Float16*)(ws + off); off += (size_t)M * D * 2;

  k_f32_to_f16<<<(M * D) / 256, 256, 0, stream>>>(x, xh, M * D);
  k_transpose_f16<<<(D * D) / 256, 256, 0, stream>>>(Wq, WqT, D);
  k_transpose_f16<<<(D * D) / 256, 256, 0, stream>>>(Wk, WkT, D);
  k_transpose_f16<<<(D * D) / 256, 256, 0, stream>>>(Wv, WvT, D);
  k_transpose_f16<<<(D * D) / 256, 256, 0, stream>>>(Wo, WoT, D);

  dim3 grid(M / 128, D / 64);
  k_gemm_proj<<<grid, 256, 0, stream>>>(xh, WqT, Qh, M, D, D, S, 0);
  k_gemm_proj<<<grid, 256, 0, stream>>>(xh, WkT, Kh, M, D, D, S, 0);
  k_gemm_proj<<<grid, 256, 0, stream>>>(xh, WvT, Vt, M, D, D, S, 1);

  k_attn<<<B * H * (S / 16), 32, 0, stream>>>(Qh, Kh, Vt, ctxh, S);

  k_gemm_out<<<grid, 256, 0, stream>>>(ctxh, WoT, bo, out, M, D, D);
}
